// ToyMoEClassifier_67276367725128
// MI455X (gfx1250) — matrix operand order repaired and checked by reading, not compiled, not run on hardware
//
#include <hip/hip_runtime.h>
#include <hip/hip_bf16.h>

// ---------------- problem constants ----------------
#define Dm   1024
#define FFm  4096
#define Em   8
#define Km   2
#define Tm   8192          // B*S = 4*2048
#define Bm   4
#define Sm   2048
#define CAP  2560          // ceil(1.25 * T * K / E)
#define NCLS 2
#define LSCALE 2.0f        // alpha/r = 4/2

typedef __bf16 bf16_t;
typedef bf16_t v16bf __attribute__((ext_vector_type(16)));
typedef float  v8f   __attribute__((ext_vector_type(8)));

struct Pack32 { uint4 lo; uint4 hi; };   // 32 bytes == v16bf

static __device__ __forceinline__ float bf2f(unsigned short u) {
    return __uint_as_float(((unsigned int)u) << 16);
}
static __device__ __forceinline__ unsigned short f2bf(float f) {
    unsigned int u = __float_as_uint(f);
    unsigned int r = u + 0x7fffu + ((u >> 16) & 1u);   // RNE
    return (unsigned short)(r >> 16);
}
static __device__ __forceinline__ float gelu_tanh(float x) {
    float x3 = x * x * x;
    return 0.5f * x * (1.0f + tanhf(0.7978845608028654f * (x + 0.044715f * x3)));
}
// 16 bf16 from two 16B chunks (A fragment: k = base + e)
static __device__ __forceinline__ v16bf load_ab2(const unsigned short* p0,
                                                 const unsigned short* p1) {
    Pack32 pk;
    pk.lo = *reinterpret_cast<const uint4*>(p0);
    pk.hi = *reinterpret_cast<const uint4*>(p1);
    return __builtin_bit_cast(v16bf, pk);
}
// 16 contiguous bf16 (B fragment: k = base + e)
static __device__ __forceinline__ v16bf load_b16(const unsigned short* p) {
    Pack32 pk;
    pk.lo = *reinterpret_cast<const uint4*>(p);
    pk.hi = *reinterpret_cast<const uint4*>(p + 8);
    return __builtin_bit_cast(v16bf, pk);
}

// ---------------- 1. gating: logits -> top2 -> softmax ----------------
__global__ __launch_bounds__(256) void k_gate(const float* __restrict__ x,
                                              const float* __restrict__ Wg,
                                              int* __restrict__ idx,
                                              float* __restrict__ gates) {
    const int lane = threadIdx.x & 31;
    const int wave = threadIdx.x >> 5;
    const int t = blockIdx.x * 8 + wave;
    if (t >= Tm) return;
    float part[Em];
    #pragma unroll
    for (int e = 0; e < Em; ++e) part[e] = 0.f;
    const float* xp = x + (size_t)t * Dm;
    for (int i = 0; i < Dm / 32; ++i) {
        int d = i * 32 + lane;
        float xv = xp[d];
        #pragma unroll
        for (int e = 0; e < Em; ++e) part[e] += xv * Wg[d * Em + e];
    }
    #pragma unroll
    for (int e = 0; e < Em; ++e)
        for (int off = 16; off > 0; off >>= 1)
            part[e] += __shfl_xor(part[e], off, 32);
    if (lane == 0) {
        float v0 = -1e30f; int e0 = 0;
        #pragma unroll
        for (int e = 0; e < Em; ++e) if (part[e] > v0) { v0 = part[e]; e0 = e; }
        float v1 = -1e30f; int e1 = 0;
        #pragma unroll
        for (int e = 0; e < Em; ++e) if (e != e0 && part[e] > v1) { v1 = part[e]; e1 = e; }
        float p1 = __expf(v1 - v0);
        float inv = 1.0f / (1.0f + p1);
        idx[t * 2 + 0] = e0;  idx[t * 2 + 1] = e1;
        gates[t * 2 + 0] = inv; gates[t * 2 + 1] = p1 * inv;
    }
}

// ---------------- 2. init dispatch-source map ----------------
__global__ __launch_bounds__(256) void k_init(int* __restrict__ disp_src, int n) {
    int i = blockIdx.x * 256 + threadIdx.x;
    if (i < n) disp_src[i] = -1;
}

// ---------------- 3. capacity scan (slot-major, deterministic) ----------------
__global__ __launch_bounds__(256) void k_scan(const int* __restrict__ idx,
                                              const float* __restrict__ gates,
                                              int* __restrict__ slot_idx,
                                              float* __restrict__ slotw,
                                              int* __restrict__ disp_src) {
    __shared__ int cnt[256][Em];
    const int tid = threadIdx.x;
    const int CH = (Km * Tm) / 256;           // 64 entries per thread
    int local[Em];
    #pragma unroll
    for (int e = 0; e < Em; ++e) local[e] = 0;
    for (int j = tid * CH; j < tid * CH + CH; ++j) {
        int s = j >> 13, t = j & (Tm - 1);
        local[idx[t * 2 + s]]++;
    }
    #pragma unroll
    for (int e = 0; e < Em; ++e) cnt[tid][e] = local[e];
    __syncthreads();
    if (tid < Em) {                            // exclusive prefix per expert
        int b = 0;
        for (int i = 0; i < 256; ++i) { int v = cnt[i][tid]; cnt[i][tid] = b; b += v; }
    }
    __syncthreads();
    int run[Em];
    #pragma unroll
    for (int e = 0; e < Em; ++e) run[e] = cnt[tid][e];
    for (int j = tid * CH; j < tid * CH + CH; ++j) {
        int s = j >> 13, t = j & (Tm - 1);
        int e = idx[t * 2 + s];
        int pos = run[e]++;
        bool keep = pos < CAP;
        int sl = e * CAP + (pos < CAP - 1 ? pos : CAP - 1);
        slot_idx[j] = sl;
        slotw[j] = keep ? gates[t * 2 + s] : 0.f;
        if (keep) disp_src[sl] = t;
    }
}

// ---------------- 4/5. weight transpose + bf16 convert ----------------
__global__ __launch_bounds__(256) void k_w1t(const float* __restrict__ W1,
                                             unsigned short* __restrict__ w1t) {
    int i = blockIdx.x * 256 + threadIdx.x;       // i = f*Dm + d
    if (i >= FFm * Dm) return;
    int f = i / Dm, d = i - f * Dm;
    w1t[i] = f2bf(W1[(size_t)d * FFm + f]);
}
__global__ __launch_bounds__(256) void k_w2t(const float* __restrict__ W2,
                                             unsigned short* __restrict__ w2t) {
    int i = blockIdx.x * 256 + threadIdx.x;       // i = d*FFm + f
    if (i >= Dm * FFm) return;
    int d = i / FFm, f = i - d * FFm;
    w2t[i] = f2bf(W2[(size_t)f * Dm + d]);
}

// ---------------- 6. dispatch tokens into bf16 capacity buffer ----------------
__global__ __launch_bounds__(256) void k_dispatch(const float* __restrict__ x,
                                                  const int* __restrict__ disp_src,
                                                  unsigned short* __restrict__ disp) {
    const int row = blockIdx.x;
    const int src = disp_src[row];
    unsigned short* dp = disp + (size_t)row * Dm;
    if (src >= 0) {
        const float* xp = x + (size_t)src * Dm;
        for (int d = threadIdx.x; d < Dm; d += 256) dp[d] = f2bf(xp[d]);
    } else {
        for (int d = threadIdx.x; d < Dm; d += 256) dp[d] = 0;
    }
}

// ---------------- 7. fused expert FFN (WMMA bf16, h tile lives in LDS) ------
__global__ __launch_bounds__(256) void k_ffn(const unsigned short* __restrict__ disp,
                                             const unsigned short* __restrict__ w1t,
                                             const unsigned short* __restrict__ w2t,
                                             const float* __restrict__ A1,
                                             const float* __restrict__ B1,
                                             const float* __restrict__ A2,
                                             const float* __restrict__ B2,
                                             float* __restrict__ y) {
    extern __shared__ unsigned char smem[];
    unsigned short* h_lds = (unsigned short*)smem;              // 16 x 4096 bf16 = 128 KB
    float* u1 = (float*)(smem + 16 * FFm * 2);                  // [16][2]
    float* u2 = u1 + 32;                                        // [16][2]

    const int tile = blockIdx.x;
    const int row0 = tile * 16;
    const int e = row0 / CAP;
    const int tid  = threadIdx.x;
    const int lane = tid & 31;
    const int wave = tid >> 5;

    if (tid < 64) u1[tid] = 0.f;
    __syncthreads();

    // u1[m][r] = sum_d disp[row0+m][d] * A1[e][d][r]   (rank-2, f32 VALU)
    {
        int pr = tid & 31, m = pr >> 1, r = pr & 1, chunk = tid >> 5;
        const unsigned short* dp = disp + (size_t)(row0 + m) * Dm;
        const float* a1 = A1 + (size_t)e * Dm * 2 + r;
        float s = 0.f;
        for (int d = chunk * 128; d < chunk * 128 + 128; ++d)
            s += bf2f(dp[d]) * a1[(size_t)d * 2];
        atomicAdd(&u1[pr], s);
    }
    __syncthreads();

    const int mA = lane & 15;
    const int kA = (lane >> 4) * 16;       // A frag: 16 contiguous
    const int nL = lane & 15;
    const int kB = (lane >> 4) * 16;       // B frag: 16 contiguous

    // ---- phase 1: h = gelu(disp @ W1 + SCALE * u1 @ B1) ----
    const float* b1p = B1 + (size_t)e * 2 * FFm;
    const float* a2p = A2 + (size_t)e * FFm * 2;
    const unsigned short* pa = disp + (size_t)(row0 + mA) * Dm + kA;
    for (int nt = wave; nt < FFm / 16; nt += 8) {
        const int n = nt * 16 + nL;
        const unsigned short* pb = w1t + (size_t)n * Dm + kB;
        v8f acc = {};
        for (int kk = 0; kk < Dm; kk += 32) {
            __builtin_prefetch(pb + kk + 128, 0, 1);
            v16bf a = load_ab2(pa + kk, pa + kk + 8);
            v16bf b = load_b16(pb + kk);
            acc = __builtin_amdgcn_wmma_f32_16x16x32_bf16(
                      false, a, false, b, (short)0, acc, false, false);
        }
        float lw0 = b1p[n], lw1 = b1p[FFm + n];
        float aw0 = a2p[(size_t)n * 2 + 0], aw1 = a2p[(size_t)n * 2 + 1];
        #pragma unroll
        for (int v = 0; v < 8; ++v) {
            int m = v + ((lane >> 4) * 8);
            float hpre = acc[v] + LSCALE * (u1[m * 2 + 0] * lw0 + u1[m * 2 + 1] * lw1);
            float hg = gelu_tanh(hpre);
            h_lds[m * FFm + n] = f2bf(hg);
            atomicAdd(&u2[m * 2 + 0], hg * aw0);
            atomicAdd(&u2[m * 2 + 1], hg * aw1);
        }
    }
    __syncthreads();

    // ---- phase 2: y = h @ W2 + SCALE * u2 @ B2  (A from LDS) ----
    const float* b2p = B2 + (size_t)e * 2 * Dm;
    for (int nt = wave; nt < Dm / 16; nt += 8) {
        const int n = nt * 16 + nL;
        const unsigned short* pb = w2t + (size_t)n * FFm + kB;
        v8f acc = {};
        for (int kk = 0; kk < FFm; kk += 32) {
            __builtin_prefetch(pb + kk + 128, 0, 1);
            v16bf a = load_ab2(h_lds + mA * FFm + kA + kk,
                               h_lds + mA * FFm + kA + kk + 8);
            v16bf b = load_b16(pb + kk);
            acc = __builtin_amdgcn_wmma_f32_16x16x32_bf16(
                      false, a, false, b, (short)0, acc, false, false);
        }
        float lw0 = b2p[n], lw1 = b2p[Dm + n];
        #pragma unroll
        for (int v = 0; v < 8; ++v) {
            int m = v + ((lane >> 4) * 8);
            float yv = acc[v] + LSCALE * (u2[m * 2 + 0] * lw0 + u2[m * 2 + 1] * lw1);
            y[(size_t)(row0 + m) * Dm + nt * 16 + nL] = yv;
        }
    }
}

// ---------------- 8. combine (gather expert outputs, weighted) --------------
__global__ __launch_bounds__(256) void k_combine(const float* __restrict__ y,
                                                 const int* __restrict__ slot_idx,
                                                 const float* __restrict__ slotw,
                                                 float* __restrict__ out_tok) {
    const int t = blockIdx.x;
    const int s0 = slot_idx[t],      s1 = slot_idx[Tm + t];
    const float w0 = slotw[t],       w1 = slotw[Tm + t];
    const float* y0 = y + (size_t)s0 * Dm;
    const float* y1 = y + (size_t)s1 * Dm;
    float* op = out_tok + (size_t)t * Dm;
    for (int d = threadIdx.x; d < Dm; d += 256)
        op[d] = y0[d] * w0 + y1[d] * w1;
}

// ---------------- 9. mean pool over sequence ----------------
__global__ __launch_bounds__(256) void k_pool(const float* __restrict__ out_tok,
                                              float* __restrict__ pooled) {
    int i = blockIdx.x * 256 + threadIdx.x;     // i = b*Dm + d
    if (i >= Bm * Dm) return;
    int b = i / Dm, d = i - b * Dm;
    float s = 0.f;
    const float* p = out_tok + ((size_t)b * Sm) * Dm + d;
    for (int t = 0; t < Sm; ++t) s += p[(size_t)t * Dm];
    pooled[i] = s * (1.0f / Sm);
}

// ---------------- 10. classifier head ----------------
__global__ __launch_bounds__(256) void k_head(const float* __restrict__ pooled,
                                              const float* __restrict__ Wh,
                                              const float* __restrict__ bh,
                                              float* __restrict__ out) {
    const int lane = threadIdx.x & 31;
    const int pair = threadIdx.x >> 5;           // 8 pairs = (b, c)
    const int b = pair >> 1, c = pair & 1;
    float s = 0.f;
    for (int d = lane; d < Dm; d += 32)
        s += pooled[b * Dm + d] * Wh[d * NCLS + c];
    for (int off = 16; off > 0; off >>= 1) s += __shfl_xor(s, off, 32);
    if (lane == 0) out[b * NCLS + c] = s + bh[c];
}

// ---------------- launcher ----------------
extern "C" void kernel_launch(void* const* d_in, const int* in_sizes, int n_in,
                              void* d_out, int out_size, void* d_ws, size_t ws_size,
                              hipStream_t stream) {
    const float* x  = (const float*)d_in[0];
    const float* Wg = (const float*)d_in[1];
    const float* W1 = (const float*)d_in[2];
    const float* W2 = (const float*)d_in[3];
    const float* A1 = (const float*)d_in[4];
    const float* B1 = (const float*)d_in[5];
    const float* A2 = (const float*)d_in[6];
    const float* B2 = (const float*)d_in[7];
    const float* Wh = (const float*)d_in[8];
    const float* bh = (const float*)d_in[9];
    float* out = (float*)d_out;

    char* base = (char*)d_ws;
    size_t off = 0;
    auto take = [&](size_t bytes) -> char* {
        char* p = base + off;
        off += (bytes + 255) & ~(size_t)255;
        return p;
    };
    int*            idx      = (int*)           take((size_t)Tm * 2 * 4);
    float*          gates    = (float*)         take((size_t)Tm * 2 * 4);
    int*            slot_idx = (int*)           take((size_t)Km * Tm * 4);
    float*          slotw    = (float*)         take((size_t)Km * Tm * 4);
    int*            disp_src = (int*)           take((size_t)Em * CAP * 4);
    unsigned short* dispb    = (unsigned short*)take((size_t)Em * CAP * Dm * 2);
    unsigned short* w1t      = (unsigned short*)take((size_t)Dm * FFm * 2);
    unsigned short* w2t      = (unsigned short*)take((size_t)Dm * FFm * 2);
    float*          yb       = (float*)         take((size_t)Em * CAP * Dm * 4);
    float*          out_tok  = (float*)         take((size_t)Tm * Dm * 4);
    float*          pooled   = (float*)         take((size_t)Bm * Dm * 4);
    (void)ws_size; (void)in_sizes; (void)n_in; (void)out_size;

    k_gate<<<Tm / 8, 256, 0, stream>>>(x, Wg, idx, gates);
    k_init<<<(Em * CAP + 255) / 256, 256, 0, stream>>>(disp_src, Em * CAP);
    k_scan<<<1, 256, 0, stream>>>(idx, gates, slot_idx, slotw, disp_src);
    k_w1t<<<(FFm * Dm + 255) / 256, 256, 0, stream>>>(W1, w1t);
    k_w2t<<<(Dm * FFm + 255) / 256, 256, 0, stream>>>(W2, w2t);
    k_dispatch<<<Em * CAP, 256, 0, stream>>>(x, disp_src, dispb);

    const size_t shmem = (size_t)16 * FFm * 2 + 64 * 4;   // h tile + u1/u2
    k_ffn<<<(Em * CAP) / 16, 256, shmem, stream>>>(dispb, w1t, w2t,
                                                   A1, B1, A2, B2, yb);

    k_combine<<<Tm, 256, 0, stream>>>(yb, slot_idx, slotw, out_tok);
    k_pool<<<(Bm * Dm + 255) / 256, 256, 0, stream>>>(out_tok, pooled);
    k_head<<<1, 256, 0, stream>>>(pooled, Wh, bh, out);
}
